// Decoder_75265006895474
// MI455X (gfx1250) — matrix operand order repaired and checked by reading, not compiled, not run on hardware
//
#include <hip/hip_runtime.h>

// ---------------- problem constants ----------------
#define T_STEPS 768
#define BATCH   32
#define NMEL    80
#define DURD    544
#define PRE     256
#define RNN     1024
#define G4      4096          // 4*RNN
#define KIN     800           // PRE + DURD
#define KA1     1824          // KIN + RNN (real K of fused [W_ih1 | W_hh1])
#define KA1P    1920          // padded to 15*128 (tail zeros)
#define KA2     2048          // RNN + RNN (fused [W_ih2 | W_hh2]) = 16*128
#define CK      128           // uniform K-chunk (32 rows x 128 bf16 = 8KB / buffer)
#define NC1     (KA1P / CK)   // 15
#define NC2     (KA2 / CK)    // 16
#define NWG     64            // persistent workgroups (co-resident)

// ---------------- workspace layout (bytes, 256-aligned) ----------------
constexpr size_t OFF_CELLIN = 0;                                              // bf16 [T][B][800]
constexpr size_t OFF_WC1    = OFF_CELLIN + (size_t)T_STEPS*BATCH*KIN*2;       // bf16 [4096][1920]
constexpr size_t OFF_WC2    = OFF_WC1    + (size_t)G4*KA1P*2;                 // bf16 [4096][2048]
constexpr size_t OFF_WPJ    = OFF_WC2    + (size_t)G4*KA2*2;                  // bf16 [80][1024]
constexpr size_t OFF_PD     = OFF_WPJ    + (size_t)NMEL*RNN*2;                // f32  [T][B][80]
constexpr size_t OFF_H1     = OFF_PD     + (size_t)T_STEPS*BATCH*NMEL*4;      // bf16 [2][B][1024]
constexpr size_t OFF_H2     = OFF_H1     + (size_t)2*BATCH*RNN*2;
constexpr size_t OFF_C1     = OFF_H2     + (size_t)2*BATCH*RNN*2;             // f32  [B][1024]
constexpr size_t OFF_C2     = OFF_C1     + (size_t)BATCH*RNN*4;
constexpr size_t OFF_BAR    = OFF_C2     + (size_t)BATCH*RNN*4;               // 2 x u32
constexpr int    INIT_WORDS = (int)((OFF_BAR + 256 - OFF_H1) / 4);            // zero span

// ---------------- WMMA types ----------------
typedef __bf16 v16bf __attribute__((ext_vector_type(16)));
typedef float  v8f   __attribute__((ext_vector_type(8)));
union AFrag { uint4 q[2]; v16bf v; };

__device__ __forceinline__ v8f vzero8() { v8f z = {0.f,0.f,0.f,0.f,0.f,0.f,0.f,0.f}; return z; }

__device__ __forceinline__ v8f wmma_bf16(v16bf a, v16bf b, v8f c) {
  return __builtin_amdgcn_wmma_f32_16x16x32_bf16(false, a, false, b, (short)0, c, false, false);
}

// A-fragment (16x32, 16-bit): lane<16 holds K=k..k+15, lane>=16 holds
// K=k+16..k+31 of row (m0+lane%16). Two 16B loads per lane.
__device__ __forceinline__ v16bf loadA(const unsigned short* base, int ldk, int row,
                                       int k, int half) {
  AFrag f;
  const char* p = (const char*)(base + (size_t)row*ldk + k + half*16);
  f.q[0] = *(const uint4*)p;
  f.q[1] = *(const uint4*)(p + 16);   // +8 elements
  return f.v;
}

// B-fragment (32x16, 16-bit): lane<16: K=k..k+15 of column n0+lane; lane>=16: K=k+16..k+31.
// Weights stored [N][K] row-major -> one contiguous 32B load per lane.
__device__ __forceinline__ v16bf loadB(const unsigned short* wrow_k) {
  return *(const v16bf*)(const void*)wrow_k;
}

// One 16B slice of the concatenated A activation: [src0 (len0) | src1 (lenTot-len0) | 0-pad]
__device__ __forceinline__ uint4 fetch2(const unsigned short* s0, int stride0, int len0,
                                        const unsigned short* s1, int stride1, int lenTot,
                                        int b, int k) {
  uint4 v = {0u, 0u, 0u, 0u};
  if (k < len0)        v = *(const uint4*)(const void*)(s0 + (size_t)b*stride0 + k);
  else if (k < lenTot) v = *(const uint4*)(const void*)(s1 + (size_t)b*stride1 + (k - len0));
  return v;
}

// ---------------- helpers ----------------
__device__ __forceinline__ unsigned short f2bf(float f) {
  unsigned u = __float_as_uint(f);
  unsigned r = u + 0x7fffu + ((u >> 16) & 1u);   // RNE
  return (unsigned short)(r >> 16);
}
__device__ __forceinline__ float u01(unsigned x, unsigned stream) {
  x = x * 0x9E3779B1u + stream;
  x ^= x >> 16; x *= 0x7feb352dU; x ^= x >> 15; x *= 0x846ca68bU; x ^= x >> 16;
  return (float)(x >> 8) * (1.0f/16777216.0f);
}
__device__ __forceinline__ float sigf(float x) { return 1.f / (1.f + __expf(-x)); }

// ---------------- grid-wide barrier (persistent kernel) ----------------
__device__ __forceinline__ void gridSync(unsigned* cnt, volatile unsigned* gen) {
  __syncthreads();
  if (threadIdx.x == 0) {
    unsigned g = *gen;
    __threadfence();
    if (atomicAdd(cnt, 1u) == NWG - 1u) {
      *cnt = 0u;
      __threadfence();
      atomicAdd((unsigned*)gen, 1u);
    } else {
      while (*gen == g) { __builtin_amdgcn_s_sleep(2); }
    }
  }
  __syncthreads();
}

// ================= prep kernels =================

__global__ void init_state(unsigned* p, int n) {
  int i = blockIdx.x * blockDim.x + threadIdx.x;
  if (i < n) p[i] = 0u;
}

// fused [W_ih1 | W_hh1 | 0-pad] -> bf16 [4096][1920]
__global__ void pack_wcat1(const float* __restrict__ wih, const float* __restrict__ whh,
                           unsigned short* __restrict__ dst) {
  int n = blockIdx.x;
  for (int k = threadIdx.x; k < KA1P; k += blockDim.x) {
    float v = (k < KIN) ? wih[(size_t)n*KIN + k]
            : (k < KA1) ? whh[(size_t)n*RNN + (k - KIN)] : 0.f;
    dst[(size_t)n*KA1P + k] = f2bf(v);
  }
}
// fused [W_ih2 | W_hh2] -> bf16 [4096][2048]
__global__ void pack_wcat2(const float* __restrict__ wih, const float* __restrict__ whh,
                           unsigned short* __restrict__ dst) {
  int n = blockIdx.x;
  for (int k = threadIdx.x; k < KA2; k += blockDim.x) {
    float v = (k < RNN) ? wih[(size_t)n*RNN + k] : whh[(size_t)n*RNN + (k - RNN)];
    dst[(size_t)n*KA2 + k] = f2bf(v);
  }
}
// W_proj[:, 0:1024] -> bf16 [80][1024]
__global__ void pack_wproj(const float* __restrict__ wp, unsigned short* __restrict__ dst) {
  int n = blockIdx.x;
  for (int k = threadIdx.x; k < RNN; k += blockDim.x)
    dst[(size_t)n*RNN + k] = f2bf(wp[(size_t)n*(RNN + DURD) + k]);
}

// Prenet (with always-on dropout) + duration concat -> cellin bf16 [T][B][800]
__global__ __launch_bounds__(256) void prenet_cellin(
    const float* __restrict__ dec,   // [B][80][T]
    const float* __restrict__ dur,   // [B][T][544]
    const float* __restrict__ W1,    // [256][80]
    const float* __restrict__ W2,    // [256][256]
    unsigned short* __restrict__ cellin) {
  __shared__ float sdi[NMEL];
  __shared__ float sy1[PRE];
  int t = blockIdx.x >> 5;
  int b = blockIdx.x & 31;
  int tid = threadIdx.x;
  if (tid < NMEL)
    sdi[tid] = (t == 0) ? 0.f : dec[((size_t)b*NMEL + tid)*T_STEPS + (t - 1)];
  __syncthreads();
  {
    float s = 0.f;
    const float* w = W1 + (size_t)tid*NMEL;
    for (int k = 0; k < NMEL; ++k) s += sdi[k] * w[k];
    s = fmaxf(s, 0.f);
    unsigned idx = (unsigned)(((unsigned)t*32u + (unsigned)b)*256u + (unsigned)tid);
    s = (u01(idx, 0xD001u) < 0.5f) ? s * 2.f : 0.f;   // inverted dropout p=0.5
    sy1[tid] = s;
  }
  __syncthreads();
  size_t row = ((size_t)t*BATCH + b) * KIN;
  {
    float s2 = 0.f;
    const float* w = W2 + (size_t)tid*PRE;
    for (int k = 0; k < PRE; ++k) s2 += sy1[k] * w[k];
    s2 = fmaxf(s2, 0.f);
    unsigned idx = (unsigned)(((unsigned)t*32u + (unsigned)b)*256u + (unsigned)tid);
    s2 = (u01(idx, 0xD002u) < 0.5f) ? s2 * 2.f : 0.f;
    cellin[row + tid] = f2bf(s2);
  }
  for (int k = tid; k < DURD; k += 256)
    cellin[row + PRE + k] = f2bf(dur[((size_t)b*T_STEPS + t)*DURD + k]);
}

// PD[t][b][n] = dur[t,b,:] @ W_proj[:,1024:].T + b_proj  (precomputed projection part)
__global__ __launch_bounds__(128) void proj_dur(
    const float* __restrict__ dur, const float* __restrict__ wp,
    const float* __restrict__ bp, float* __restrict__ pd) {
  __shared__ float sd[DURD];
  int t = blockIdx.x >> 5, b = blockIdx.x & 31, tid = threadIdx.x;
  for (int k = tid; k < DURD; k += 128)
    sd[k] = dur[((size_t)b*T_STEPS + t)*DURD + k];
  __syncthreads();
  if (tid < NMEL) {
    float s = bp[tid];
    const float* w = wp + (size_t)tid*(RNN + DURD) + RNN;
    for (int k = 0; k < DURD; ++k) s += sd[k] * w[k];
    pd[((size_t)t*BATCH + b)*NMEL + tid] = s;
  }
}

// ================= persistent recurrent kernel =================
// 64 WGs x 256 threads (8 wave32). WG wg owns units [wg*16, wg*16+16).
// Wave w computes tile: gate g=w>>1 (i,f,g,o), batch half m=w&1 -> cell update is WG-local.
__global__ __launch_bounds__(256) void decoder_recurrent(
    const unsigned short* __restrict__ cellin,
    const unsigned short* __restrict__ Wc1,
    const unsigned short* __restrict__ Wc2,
    const unsigned short* __restrict__ Wpj,
    const float* __restrict__ pd,
    const float* __restrict__ bih1, const float* __restrict__ bhh1,
    const float* __restrict__ bih2, const float* __restrict__ bhh2,
    unsigned short* __restrict__ h1b,   // [2][B][1024] bf16, parity double-buffer
    unsigned short* __restrict__ h2b,
    float* __restrict__ c1, float* __restrict__ c2,
    unsigned* __restrict__ bar,
    float* __restrict__ out)            // [B][80][T]
{
  __shared__ __align__(32) unsigned short sA[2 * BATCH * CK]; // ping-pong, 2x8KB
  __shared__ float gbuf[4 * BATCH * 16];                      // gates i,f,g,o of owned units
  __shared__ float hbuf[BATCH * 16];                          // new h (pre-zoneout)

  const int tid  = threadIdx.x;
  const int wg   = blockIdx.x;
  const int wave = tid >> 5;
  const int lane = tid & 31;
  const int half = lane >> 4;
  const int lr   = lane & 15;
  const int gI   = wave >> 1;          // gate 0..3
  const int m0   = (wave & 1) << 4;    // batch tile base
  const int j0   = wg << 4;            // unit base
  const int n0g  = gI * RNN + j0;      // gate-row base in [4096]

  // staging role: thread copies two 16B quads of row bS each chunk (shift/mask only)
  const int bS = tid >> 3;             // 0..31
  const int qS = tid & 7;              // quad 0..7 (covers q and q+8)

  // per-unit bias sums are step-invariant -> hoist out of the t loop (register values,
  // unaffected by the per-step memory barrier below)
  float b1[2][4], b2[2][4];
  #pragma unroll
  for (int it = 0; it < 2; ++it) {
    int unit = j0 + ((tid + (it << 8)) & 15);
    #pragma unroll
    for (int g = 0; g < 4; ++g) {
      b1[it][g] = bih1[g*RNN + unit] + bhh1[g*RNN + unit];
      b2[it][g] = bih2[g*RNN + unit] + bhh2[g*RNN + unit];
    }
  }

  unsigned* cnt = bar;
  volatile unsigned* gen = bar + 1;

  for (int t = 0; t < T_STEPS; ++t) {
    // Block LICM across time steps: without this, the compiler hoists the
    // t-invariant weight-fragment loads out of the loop, blows past 256 VGPRs
    // (s_set_vgpr_msb churn) and spills them to scratch, turning the inner loop's
    // L2 global_load_b128 stream into scratch reload traffic. Weights are
    // L2-resident anyway, so re-loading per step is the cheap option.
    asm volatile("" ::: "memory");

    const int wp = t & 1, rp = wp ^ 1;
    const unsigned short* h1r = h1b + (size_t)rp * BATCH * RNN;   // h1 from t-1
    const unsigned short* h2r = h2b + (size_t)rp * BATCH * RNN;   // h2 from t-1
    unsigned short* h1w = h1b + (size_t)wp * BATCH * RNN;
    unsigned short* h2w = h2b + (size_t)wp * BATCH * RNN;
    const unsigned short* cin = cellin + (size_t)t * BATCH * KIN;

    // ---------------- GEMM1: gates1 = [cellin | h1_prev | 0] x Wc1^T ----------------
    v8f acc = vzero8();
    const unsigned short* wrow = Wc1 + (size_t)(n0g + lr) * KA1P;
    // prologue: stage chunk 0 into buffer 0
    #pragma unroll
    for (int qq = 0; qq < 2; ++qq) {
      int q = qS + qq*8;
      *(uint4*)(void*)(sA + (size_t)bS*CK + q*8) =
          fetch2(cin, KIN, KIN, h1r, RNN, KA1, bS, q*8);
    }
    __syncthreads();
    for (int c = 0; c < NC1; ++c) {
      const int p = c & 1;
      if (c + 1 < NC1) {                       // stage chunk c+1 into the other buffer
        #pragma unroll
        for (int qq = 0; qq < 2; ++qq) {
          int q = qS + qq*8;
          *(uint4*)(void*)(sA + (p^1)*(BATCH*CK) + (size_t)bS*CK + q*8) =
              fetch2(cin, KIN, KIN, h1r, RNN, KA1, bS, (c+1)*CK + q*8);
        }
      }
      const unsigned short* abase = sA + p*(BATCH*CK);
      #pragma unroll
      for (int ks = 0; ks < CK; ks += 32) {
        v16bf a  = loadA(abase, CK, m0 + lr, ks, half);
        v16bf bb = loadB(wrow + c*CK + ks + half*16);
        acc = wmma_bf16(a, bb, acc);
      }
      __syncthreads();
    }
    #pragma unroll
    for (int r = 0; r < 8; ++r) {              // D layout: lane<16 M=r, lane>=16 M=r+8
      int b = m0 + half*8 + r;
      gbuf[(gI*BATCH + b)*16 + lr] = acc[r];
    }
    __syncthreads();
    // cell1 update (owned units only, WG-local)
    #pragma unroll
    for (int it = 0; it < 2; ++it) {
      int idx = tid + (it << 8);               // 0..511 = (b,jj)
      int b = idx >> 4, jj = idx & 15, unit = j0 + jj;
      float gi = gbuf[(0*BATCH + b)*16 + jj] + b1[it][0];
      float gf = gbuf[(1*BATCH + b)*16 + jj] + b1[it][1];
      float gg = gbuf[(2*BATCH + b)*16 + jj] + b1[it][2];
      float go = gbuf[(3*BATCH + b)*16 + jj] + b1[it][3];
      float c  = c1[(size_t)b*RNN + unit];
      float cn = sigf(gf)*c + sigf(gi)*tanhf(gg);
      c1[(size_t)b*RNN + unit] = cn;
      hbuf[idx] = sigf(go)*tanhf(cn);
    }
    __syncthreads();
    #pragma unroll
    for (int it = 0; it < 2; ++it) {           // zoneout (row-0 broadcast bug kept)
      int idx = tid + (it << 8);
      int b = idx >> 4, jj = idx & 15, unit = j0 + jj;
      float z = (u01((unsigned)(t*RNN + unit), 0x5A01u) < 0.1f) ? 1.f : 0.f;
      float h = z * hbuf[jj] + (1.f - z) * hbuf[idx];
      h1w[(size_t)b*RNN + unit] = f2bf(h);
    }
    gridSync(cnt, gen);   // h1(t) globally visible

    // ---------------- GEMM2: gates2 = [h1 | h2_prev] x Wc2^T ----------------
    acc = vzero8();
    const unsigned short* wrow2 = Wc2 + (size_t)(n0g + lr) * KA2;
    const unsigned short* h1c = h1b + (size_t)wp * BATCH * RNN;
    #pragma unroll
    for (int qq = 0; qq < 2; ++qq) {
      int q = qS + qq*8;
      *(uint4*)(void*)(sA + (size_t)bS*CK + q*8) =
          fetch2(h1c, RNN, RNN, h2r, RNN, KA2, bS, q*8);
    }
    __syncthreads();
    for (int c = 0; c < NC2; ++c) {
      const int p = c & 1;
      if (c + 1 < NC2) {
        #pragma unroll
        for (int qq = 0; qq < 2; ++qq) {
          int q = qS + qq*8;
          *(uint4*)(void*)(sA + (p^1)*(BATCH*CK) + (size_t)bS*CK + q*8) =
              fetch2(h1c, RNN, RNN, h2r, RNN, KA2, bS, (c+1)*CK + q*8);
        }
      }
      const unsigned short* abase = sA + p*(BATCH*CK);
      #pragma unroll
      for (int ks = 0; ks < CK; ks += 32) {
        v16bf a  = loadA(abase, CK, m0 + lr, ks, half);
        v16bf bb = loadB(wrow2 + c*CK + ks + half*16);
        acc = wmma_bf16(a, bb, acc);
      }
      __syncthreads();
    }
    #pragma unroll
    for (int r = 0; r < 8; ++r) {
      int b = m0 + half*8 + r;
      gbuf[(gI*BATCH + b)*16 + lr] = acc[r];
    }
    __syncthreads();
    #pragma unroll
    for (int it = 0; it < 2; ++it) {
      int idx = tid + (it << 8);
      int b = idx >> 4, jj = idx & 15, unit = j0 + jj;
      float gi = gbuf[(0*BATCH + b)*16 + jj] + b2[it][0];
      float gf = gbuf[(1*BATCH + b)*16 + jj] + b2[it][1];
      float gg = gbuf[(2*BATCH + b)*16 + jj] + b2[it][2];
      float go = gbuf[(3*BATCH + b)*16 + jj] + b2[it][3];
      float c  = c2[(size_t)b*RNN + unit];
      float cn = sigf(gf)*c + sigf(gi)*tanhf(gg);
      c2[(size_t)b*RNN + unit] = cn;
      hbuf[idx] = sigf(go)*tanhf(cn);
    }
    __syncthreads();
    #pragma unroll
    for (int it = 0; it < 2; ++it) {
      int idx = tid + (it << 8);
      int b = idx >> 4, jj = idx & 15, unit = j0 + jj;
      float z = (u01((unsigned)(t*RNN + unit), 0x5A02u) < 0.1f) ? 1.f : 0.f;
      float h = z * hbuf[jj] + (1.f - z) * hbuf[idx];
      h2w[(size_t)b*RNN + unit] = f2bf(h);
    }
    gridSync(cnt, gen);   // h2(t) globally visible

    // ---------------- projection: out[t] = h2 @ Wprojh^T + PD[t] ----------------
    // 10 tiles (2 batch x 5 mel) handled by the first 10 waves of the grid.
    int fw = wg*8 + wave;
    if (fw < 10) {
      int n0 = (fw >> 1) << 4;
      int m0p = (fw & 1) << 4;
      const unsigned short* h2c = h2b + (size_t)wp * BATCH * RNN;
      const unsigned short* wpr = Wpj + (size_t)(n0 + lr) * RNN;
      v8f pacc = vzero8();
      #pragma unroll 4
      for (int k = 0; k < RNN; k += 32) {
        v16bf a  = loadA(h2c, RNN, m0p + lr, k, half);
        v16bf bb = loadB(wpr + k + half*16);
        pacc = wmma_bf16(a, bb, pacc);
      }
      #pragma unroll
      for (int r = 0; r < 8; ++r) {
        int b = m0p + half*8 + r;
        int col = n0 + lr;
        float v = pacc[r] + pd[((size_t)t*BATCH + b)*NMEL + col];
        out[((size_t)b*NMEL + col)*T_STEPS + t] = v;
      }
    }
    // No end-of-step barrier needed: parity double-buffering + the two barriers
    // above already order every cross-WG read/write pair.
  }
}

// ================= host launcher =================
extern "C" void kernel_launch(void* const* d_in, const int* in_sizes, int n_in,
                              void* d_out, int out_size, void* d_ws, size_t ws_size,
                              hipStream_t stream) {
  (void)in_sizes; (void)n_in; (void)out_size; (void)ws_size;
  const float* dur   = (const float*)d_in[0];   // [B][T][544]
  const float* dec   = (const float*)d_in[1];   // [B][80][T]
  // d_in[2] memory_lengths: unused by the reference forward
  const float* W1    = (const float*)d_in[3];
  const float* W2    = (const float*)d_in[4];
  const float* Wih1  = (const float*)d_in[5];
  const float* Whh1  = (const float*)d_in[6];
  const float* bih1  = (const float*)d_in[7];
  const float* bhh1  = (const float*)d_in[8];
  const float* Wih2  = (const float*)d_in[9];
  const float* Whh2  = (const float*)d_in[10];
  const float* bih2  = (const float*)d_in[11];
  const float* bhh2  = (const float*)d_in[12];
  const float* Wproj = (const float*)d_in[13];
  const float* bproj = (const float*)d_in[14];
  float* out = (float*)d_out;

  char* ws = (char*)d_ws;
  unsigned short* cellin = (unsigned short*)(ws + OFF_CELLIN);
  unsigned short* wc1    = (unsigned short*)(ws + OFF_WC1);
  unsigned short* wc2    = (unsigned short*)(ws + OFF_WC2);
  unsigned short* wpj    = (unsigned short*)(ws + OFF_WPJ);
  float*          pdbuf  = (float*)(ws + OFF_PD);
  unsigned short* h1b    = (unsigned short*)(ws + OFF_H1);
  unsigned short* h2b    = (unsigned short*)(ws + OFF_H2);
  float*          c1     = (float*)(ws + OFF_C1);
  float*          c2     = (float*)(ws + OFF_C2);
  unsigned*       bar    = (unsigned*)(ws + OFF_BAR);

  // zero h/c state + barrier counters every call (deterministic launches)
  init_state<<<(INIT_WORDS + 255)/256, 256, 0, stream>>>((unsigned*)(ws + OFF_H1), INIT_WORDS);

  // weight packing (bf16, fused layouts; resident in L2 afterwards)
  pack_wcat1<<<G4, 256, 0, stream>>>(Wih1, Whh1, wc1);
  pack_wcat2<<<G4, 256, 0, stream>>>(Wih2, Whh2, wc2);
  pack_wproj<<<NMEL, 256, 0, stream>>>(Wproj, wpj);

  // feed-forward precompute
  prenet_cellin<<<T_STEPS*BATCH, 256, 0, stream>>>(dec, dur, W1, W2, cellin);
  proj_dur<<<T_STEPS*BATCH, 128, 0, stream>>>(dur, Wproj, bproj, pdbuf);

  // persistent recurrence
  decoder_recurrent<<<NWG, 256, 0, stream>>>(cellin, wc1, wc2, wpj, pdbuf,
                                             bih1, bhh1, bih2, bhh2,
                                             h1b, h2b, c1, c2, bar, out);
}
